// RelativeTransformerEncoder_24223615549917
// MI455X (gfx1250) — matrix operand order repaired and checked by reading, not compiled, not run on hardware
//
#include <hip/hip_runtime.h>

// ---------------- problem constants ----------------
#define B_  2
#define S_  1024
#define D_  768
#define H_  12
#define DH_ 64
#define L_  6
#define F_  3072
#define R_  32
#define M_  (B_*S_)   // 2048 rows

// ---------------- WMMA types / helpers ----------------
typedef __attribute__((ext_vector_type(16))) __bf16 v16bf;
typedef __attribute__((ext_vector_type(8)))  float  v8f;
typedef __attribute__((ext_vector_type(4)))  unsigned uvec4;
typedef __attribute__((ext_vector_type(8)))  unsigned uvec8;

struct __align__(16) U4 { unsigned int a, b, c, d; };
union FragU { v16bf v; U4 q[2]; };

__device__ __forceinline__ unsigned short f2bf(float x) {
  unsigned u = __float_as_uint(x);
  return (unsigned short)((u + 0x7FFFu + ((u >> 16) & 1u)) >> 16);  // RNE
}
__device__ __forceinline__ float bf2f(unsigned short h) {
  return __uint_as_float(((unsigned)h) << 16);
}

// A-fragment (16x32 bf16): lane<16 holds row r, K {0..7,16..23}; lane>=16 K {8..15,24..31}.
// caller passes p = row_base + 8*(lane>>4); we read p[0..7] and p[16..23].
__device__ __forceinline__ v16bf frag_a(const unsigned short* p) {
  FragU f; f.q[0] = *(const U4*)p; f.q[1] = *(const U4*)(p + 16); return f.v;
}
// B-fragment (32x16 bf16): lane<16 holds col r, K {0..7,16..23}; lane>=16 K {8..15,24..31}.
// caller passes p = col_base + 8*(lane>>4); we read p[0..7] and p[16..23].
__device__ __forceinline__ v16bf frag_b(const unsigned short* p) {
  FragU f; f.q[0] = *(const U4*)p; f.q[1] = *(const U4*)(p + 16); return f.v;
}
__device__ __forceinline__ v8f wmma_bf16(v16bf a, v16bf b, v8f c) {
  return __builtin_amdgcn_wmma_f32_16x16x32_bf16(false, a, false, b, (short)0, c, false, false);
}

// ---------------- CDNA5 async DMA: global -> LDS, 16B per lane (ASYNCcnt) ----------------
__device__ __forceinline__ void async_ld16(void* lds_ptr, const void* gptr) {
  unsigned lo = (unsigned)(uintptr_t)lds_ptr;                 // generic LDS addr[31:0] == LDS offset
  unsigned long long ga = (unsigned long long)(uintptr_t)gptr;
  asm volatile("global_load_async_to_lds_b128 %0, %1, off"
               :: "v"(lo), "v"(ga) : "memory");
}
__device__ __forceinline__ void wait_async0() {
  asm volatile("s_wait_asynccnt 0x0" ::: "memory");
}

// ---------------- CDNA5 Tensor Data Mover: 2D tile global -> LDS (TENSORcnt) ----------------
// Loads a [64 rows x 32 elems] bf16 tile (row stride = K elements) to LDS in one DMA.
// D# group0: count=1 | lds_addr | global_addr[56:0] | type=2.  group1: data_size=2B,
// tensor_dim0=K, tensor_dim1=Nrows, tile_dim0=32, tile_dim1=64, tensor_dim0_stride=K.
__device__ __forceinline__ void tdm_load_tile(unsigned lds_off, const void* gptr,
                                              int K, int Nrows) {
  unsigned long long ga = (unsigned long long)(uintptr_t)gptr;
  uvec4 g0; uvec8 g1;
  g0[0] = 1u;                                       // count=1, user descriptor
  g0[1] = lds_off;                                  // lds_addr (bytes)
  g0[2] = (unsigned)ga;                             // global_addr[31:0]
  g0[3] = ((unsigned)(ga >> 32) & 0x01FFFFFFu) | 0x80000000u;  // addr[56:32] | type=2
  g1[0] = 0x00010000u;                              // workgroup_mask=0, data_size=1 (2B)
  g1[1] = ((unsigned)K & 0xFFFFu) << 16;            // tensor_dim0[15:0] @ bits 63:48
  g1[2] = ((unsigned)K >> 16) | (((unsigned)Nrows & 0xFFFFu) << 16);  // dim0 hi | dim1 lo
  g1[3] = ((unsigned)Nrows >> 16) | (32u << 16);    // dim1 hi | tile_dim0=32
  g1[4] = 64u;                                      // tile_dim1=64, tile_dim2=0
  g1[5] = (unsigned)K;                              // tensor_dim0_stride[31:0] (elements)
  g1[6] = 0u;                                       // stride hi | tensor_dim1_stride lo
  g1[7] = 0u;
  asm volatile("tensor_load_to_lds %0, %1, null, null"
               :: "s"(g0), "s"(g1) : "memory");
}
__device__ __forceinline__ void wait_tensor0() {
  asm volatile("s_wait_tensorcnt 0x0" ::: "memory");
}

// ---------------- embeddings (analytic sinusoidal table) ----------------
__global__ void embed_kernel(const int* __restrict__ tok, const int* __restrict__ seg,
                             const int* __restrict__ pos, const int* __restrict__ qm,
                             const float* __restrict__ tok_emb, const float* __restrict__ seg_emb,
                             float* __restrict__ x) {
  int idx = blockIdx.x * 256 + threadIdx.x;           // < M_*D_
  int d = idx % D_;
  int row = idx / D_;
  int t = tok[row], sg = seg[row], q = qm[row];
  int pid = pos[row] * q;
  float di = (float)(d & ~1);
  float ang = (float)pid * __powf(10000.0f, -di / (float)D_);
  float pv = (d & 1) ? __cosf(ang) : __sinf(ang);
  x[idx] = tok_emb[(size_t)t * D_ + d] + seg_emb[(size_t)sg * D_ + d] + pv * (float)q;
}

// ---------------- LayerNorm -> bf16 activations ----------------
__global__ void ln_kernel(const float* __restrict__ x, const float* __restrict__ g,
                          const float* __restrict__ bt, unsigned short* __restrict__ out) {
  __shared__ float sb[256];
  int row = blockIdx.x, tid = threadIdx.x;
  const float* xr = x + (size_t)row * D_;
  float v0 = xr[tid], v1 = xr[tid + 256], v2 = xr[tid + 512];
  sb[tid] = v0 + v1 + v2;
  __syncthreads();
  for (int st = 128; st > 0; st >>= 1) { if (tid < st) sb[tid] += sb[tid + st]; __syncthreads(); }
  float mu = sb[0] * (1.0f / (float)D_);
  __syncthreads();
  float d0 = v0 - mu, d1 = v1 - mu, d2 = v2 - mu;
  sb[tid] = d0 * d0 + d1 * d1 + d2 * d2;
  __syncthreads();
  for (int st = 128; st > 0; st >>= 1) { if (tid < st) sb[tid] += sb[tid + st]; __syncthreads(); }
  float rstd = rsqrtf(sb[0] * (1.0f / (float)D_) + 1e-12f);
  unsigned short* orow = out + (size_t)row * D_;
  orow[tid]       = f2bf(d0 * rstd * g[tid]       + bt[tid]);
  orow[tid + 256] = f2bf(d1 * rstd * g[tid + 256] + bt[tid + 256]);
  orow[tid + 512] = f2bf(d2 * rstd * g[tid + 512] + bt[tid + 512]);
}

// ---------------- weight convert + transpose: W[K,N] f32 -> Wt[N,K] bf16 ----------------
__global__ void convT_kernel(const float* __restrict__ W, unsigned short* __restrict__ Wt,
                             int K, int N) {
  int idx = blockIdx.x * 256 + threadIdx.x;
  if (idx >= K * N) return;
  int n = idx % N, k = idx / N;
  Wt[(size_t)n * K + k] = f2bf(W[idx]);
}

// ---------------- V transpose: v[B,S,H,DH] -> vT[B,H,DH,S] (bf16) ----------------
__global__ void vT_kernel(const unsigned short* __restrict__ v, unsigned short* __restrict__ vt) {
  int idx = blockIdx.x * 256 + threadIdx.x;           // < B_*H_*DH_*S_
  int j = idx & (S_ - 1);
  int d = (idx >> 10) & (DH_ - 1);
  int hb = idx >> 16;                                 // b*H_+h
  int h = hb % H_, b = hb / H_;
  vt[idx] = v[(((size_t)(b * S_ + j)) * H_ + h) * DH_ + d];
}

// ---------------- rel bias precompute: rel[b,h,i,r] = q . rel_emb[r,h,:] ----------------
__global__ void rel_kernel(const unsigned short* __restrict__ q, const float* __restrict__ re,
                           float* __restrict__ out) {
  int idx = blockIdx.x * 256 + threadIdx.x;           // < B_*H_*S_*R_
  int r = idx & (R_ - 1);
  int i = (idx >> 5) & (S_ - 1);
  int hb = idx >> 15;                                 // b*H_+h
  int h = hb % H_, b = hb / H_;
  const unsigned short* qp = q + (((size_t)(b * S_ + i)) * H_ + h) * DH_;
  const float* rp = re + ((size_t)r * H_ + h) * DH_;
  float acc = 0.f;
  #pragma unroll 16
  for (int d = 0; d < DH_; ++d) acc += bf2f(qp[d]) * rp[d];
  out[idx] = acc;
}

// ---------------- bf16 WMMA GEMM: async-LDS A + TDM B, double-buffered ----------------
// Block (8 waves) computes a 128x64 tile of out = epilogue(A[M,K] @ Wt[N,K]^T + bias).
// Per K=32 stage: A-tile 128x32 via per-lane global_load_async_to_lds_b128 (ASYNCcnt);
// B-tile 64x32 via ONE tensor_load_to_lds DMA issued by wave 0 (TENSORcnt).
__global__ void gemm_kernel(const unsigned short* __restrict__ A,
                            const unsigned short* __restrict__ Wt,
                            const float* __restrict__ bias,
                            const float* __restrict__ resid,
                            float* __restrict__ outF,
                            unsigned short* __restrict__ outB,
                            int M, int N, int K, int doGelu) {
  __shared__ __align__(16) unsigned short Abuf[2][128][32];   // 16 KB
  __shared__ __align__(16) unsigned short Bbuf[2][64][32];    //  8 KB
  int tid = threadIdx.x;
  int lane = tid & 31, widx = tid >> 5;
  int tilesN = N >> 6;
  int tm = blockIdx.x / tilesN;
  int tn = blockIdx.x - tm * tilesN;
  int m0 = tm << 7, n0 = tn << 6;
  int r = lane & 15, hh = lane >> 4;

  // per-thread async slots for A: 2 x 16B per stage (512 xfers / 256 threads)
  int arow = tid >> 2;                 // 0..63
  int aseg = (tid & 3) * 8;            // bf16 offset within 32-wide K slice

  auto issueA = [&](int buf, int k0) {
    async_ld16(&Abuf[buf][arow][aseg],      A + (size_t)(m0 + arow) * K + k0 + aseg);
    async_ld16(&Abuf[buf][arow + 64][aseg], A + (size_t)(m0 + arow + 64) * K + k0 + aseg);
  };
  auto issueB = [&](int buf, int k0) {
    if (widx == 0)
      tdm_load_tile((unsigned)(uintptr_t)&Bbuf[buf][0][0],
                    Wt + (size_t)n0 * K + k0, K, N);
  };

  v8f acc[4];
  #pragma unroll
  for (int j = 0; j < 4; ++j)
    #pragma unroll
    for (int g = 0; g < 8; ++g) acc[j][g] = 0.f;

  issueA(0, 0);
  issueB(0, 0);
  int NK = K >> 5;
  for (int kt = 0; kt < NK; ++kt) {
    int cur = kt & 1;
    wait_async0();                      // my A transfers for stage kt done
    if (widx == 0) wait_tensor0();      // B tile DMA for stage kt done
    __syncthreads();                    // buf[cur] valid for all waves
    if (kt + 1 < NK) { issueA(cur ^ 1, (kt + 1) << 5); issueB(cur ^ 1, (kt + 1) << 5); }
    v16bf af = frag_a(&Abuf[cur][widx * 16 + r][8 * hh]);
    #pragma unroll
    for (int j = 0; j < 4; ++j) {
      v16bf bf = frag_b(&Bbuf[cur][16 * j + r][8 * hh]);
      acc[j] = wmma_bf16(af, bf, acc[j]);
    }
    __syncthreads();                    // all reads of buf[cur] done before it is refilled
  }

  #pragma unroll
  for (int j = 0; j < 4; ++j)
    #pragma unroll
    for (int g = 0; g < 8; ++g) {
      int row = m0 + widx * 16 + g + 8 * hh;          // C layout: M = g + 8*(lane/16)
      int col = n0 + 16 * j + r;                      // N = lane%16
      float v = acc[j][g] + bias[col];
      if (doGelu) {
        float u = v;
        v = 0.5f * u * (1.0f + tanhf(0.7978845608028654f * (u + 0.044715f * u * u * u)));
      }
      size_t off = (size_t)row * N + col;
      if (outF) {
        float rv = resid ? resid[off] : 0.f;
        outF[off] = rv + v;
      } else {
        outB[off] = f2bf(v);
      }
    }
}

// ---------------- flash attention with relative bias (per 16-row query tile) ----------------
__global__ void attn_kernel(const unsigned short* __restrict__ q,
                            const unsigned short* __restrict__ k,
                            const unsigned short* __restrict__ vt,
                            const float* __restrict__ rel,
                            const int* __restrict__ ids,
                            const int* __restrict__ amask,
                            unsigned short* __restrict__ ctx) {
  __shared__ __align__(16) unsigned short lp[8][512];   // 16x32 bf16 probs per wave
  int lane = threadIdx.x & 31;
  int widx = threadIdx.x >> 5;
  int t = blockIdx.x * 8 + widx;
  if (t >= B_ * H_ * (S_ / 16)) return;
  int b  = t / (H_ * (S_ / 16));
  int rm = t % (H_ * (S_ / 16));
  int h  = rm / (S_ / 16);
  int i0 = (rm % (S_ / 16)) * 16;
  int r = lane & 15, hh = lane >> 4;

  const unsigned short* qbase = q + (((size_t)(b * S_ + i0 + r)) * H_ + h) * DH_;
  v16bf qf0 = frag_a(qbase + 8 * hh);                  // d = 0..31
  v16bf qf1 = frag_a(qbase + 32 + 8 * hh);             // d = 32..63

  float M8[8], L8[8];
  v8f o[4];
  #pragma unroll
  for (int g = 0; g < 8; ++g) { M8[g] = -3.0e38f; L8[g] = 0.f; }
  #pragma unroll
  for (int j = 0; j < 4; ++j)
    #pragma unroll
    for (int g = 0; g < 8; ++g) o[j][g] = 0.f;

  unsigned short* lrow = lp[widx];
  const float sc = 0.125f;                             // 1/sqrt(64)

  for (int jt = 0; jt < S_ / 32; ++jt) {
    v8f s[2];
    #pragma unroll
    for (int p = 0; p < 2; ++p) {
      #pragma unroll
      for (int g = 0; g < 8; ++g) s[p][g] = 0.f;
      int j0 = jt * 32 + p * 16;
      const unsigned short* kb = k + (((size_t)(b * S_ + j0 + r)) * H_ + h) * DH_;
      s[p] = wmma_bf16(qf0, frag_b(kb + 8 * hh), s[p]);
      s[p] = wmma_bf16(qf1, frag_b(kb + 32 + 8 * hh), s[p]);
    }
    // relative bias gather + scale + mask (per element of C-layout)
    #pragma unroll
    for (int p = 0; p < 2; ++p) {
      int j0 = jt * 32 + p * 16;
      #pragma unroll
      for (int g = 0; g < 8; ++g) {
        int i = i0 + g + 8 * hh;
        int j = j0 + r;
        size_t ij = ((size_t)(b * S_ + i)) * S_ + j;
        int id = ids[ij];
        float sv = s[p][g] + rel[(((size_t)(b * H_ + h)) * S_ + i) * R_ + id];
        s[p][g] = sv * sc + (1.0f - (float)amask[ij]) * -10000.0f;
      }
    }
    // online softmax: rows live in half-waves; xor-shuffles 1/2/4/8 stay within halves
    #pragma unroll
    for (int g = 0; g < 8; ++g) {
      float tm = fmaxf(s[0][g], s[1][g]);
      tm = fmaxf(tm, __shfl_xor(tm, 1));
      tm = fmaxf(tm, __shfl_xor(tm, 2));
      tm = fmaxf(tm, __shfl_xor(tm, 4));
      tm = fmaxf(tm, __shfl_xor(tm, 8));
      float Mn = fmaxf(M8[g], tm);
      float alpha = __expf(M8[g] - Mn);
      float p0 = __expf(s[0][g] - Mn);
      float p1 = __expf(s[1][g] - Mn);
      s[0][g] = p0; s[1][g] = p1;
      float ts = p0 + p1;
      ts += __shfl_xor(ts, 1);
      ts += __shfl_xor(ts, 2);
      ts += __shfl_xor(ts, 4);
      ts += __shfl_xor(ts, 8);
      L8[g] = L8[g] * alpha + ts;
      M8[g] = Mn;
      o[0][g] *= alpha; o[1][g] *= alpha; o[2][g] *= alpha; o[3][g] *= alpha;
    }
    // stage probs through LDS: C-layout -> row-major 16x32 -> A-fragment
    #pragma unroll
    for (int p = 0; p < 2; ++p)
      #pragma unroll
      for (int g = 0; g < 8; ++g)
        lrow[(g + 8 * hh) * 32 + p * 16 + r] = f2bf(s[p][g]);
    // same-wave LDS ops are in-order (DScnt); load sees the stores above
    v16bf pf = frag_a(lrow + r * 32 + 8 * hh);
    #pragma unroll
    for (int jn = 0; jn < 4; ++jn) {
      const unsigned short* vb =
          vt + (((size_t)(b * H_ + h)) * DH_ + jn * 16 + r) * S_ + jt * 32 + 8 * hh;
      o[jn] = wmma_bf16(pf, frag_b(vb), o[jn]);
    }
  }
  #pragma unroll
  for (int jn = 0; jn < 4; ++jn)
    #pragma unroll
    for (int g = 0; g < 8; ++g) {
      int i = i0 + g + 8 * hh;
      float val = o[jn][g] / L8[g];
      ctx[(((size_t)(b * S_ + i)) * H_ + h) * DH_ + jn * 16 + r] = f2bf(val);
    }
}

// ---------------- host orchestration ----------------
extern "C" void kernel_launch(void* const* d_in, const int* in_sizes, int n_in,
                              void* d_out, int out_size, void* d_ws, size_t ws_size,
                              hipStream_t stream) {
  (void)in_sizes; (void)n_in; (void)out_size; (void)ws_size;
  const int*   tok     = (const int*)d_in[0];
  const int*   seg     = (const int*)d_in[1];
  const int*   pos     = (const int*)d_in[2];
  const int*   qmask   = (const int*)d_in[3];
  const int*   amask   = (const int*)d_in[4];
  const int*   relids  = (const int*)d_in[5];
  const float* tok_emb = (const float*)d_in[6];
  const float* seg_emb = (const float*)d_in[7];
  const float* ln1g    = (const float*)d_in[8];
  const float* ln1b    = (const float*)d_in[9];
  const float* ln2g    = (const float*)d_in[10];
  const float* ln2b    = (const float*)d_in[11];
  const float* wq      = (const float*)d_in[12];
  const float* bq      = (const float*)d_in[13];
  const float* wk      = (const float*)d_in[14];
  const float* bk      = (const float*)d_in[15];
  const float* wv      = (const float*)d_in[16];
  const float* bv      = (const float*)d_in[17];
  const float* wo      = (const float*)d_in[18];
  const float* bo      = (const float*)d_in[19];
  const float* rel_emb = (const float*)d_in[20];
  const float* w1      = (const float*)d_in[21];
  const float* b1      = (const float*)d_in[22];
  const float* w2      = (const float*)d_in[23];
  const float* b2      = (const float*)d_in[24];

  float* x = (float*)d_out;                            // fp32 residual stream [B,S,D]
  char* ws = (char*)d_ws;
  // workspace layout (bytes)
  unsigned short* wt   = (unsigned short*)(ws);                    //  4,718,592  (max D*F bf16)
  unsigned short* hbf  = (unsigned short*)(ws + 4718592);          //  3,145,728
  unsigned short* qb   = (unsigned short*)(ws + 7864320);          //  3,145,728
  unsigned short* kb   = (unsigned short*)(ws + 11010048);         //  3,145,728
  unsigned short* vb   = (unsigned short*)(ws + 14155776);         //  3,145,728
  unsigned short* vtb  = (unsigned short*)(ws + 17301504);         //  3,145,728
  unsigned short* ctxb = (unsigned short*)(ws + 20447232);         //  3,145,728
  unsigned short* a1b  = (unsigned short*)(ws + 23592960);         // 12,582,912
  float*          relb = (float*)(ws + 36175872);                  //  3,145,728  (end ~39.3 MB)

  embed_kernel<<<(M_ * D_) / 256, 256, 0, stream>>>(tok, seg, pos, qmask, tok_emb, seg_emb, x);

  const int gemmBlocksD = (M_ / 128) * (D_ / 64);        // 192
  const int gemmBlocksF = (M_ / 128) * (F_ / 64);        // 768
  const int convBlocksDD = (D_ * D_) / 256;              // 2304
  const int convBlocksDF = (D_ * F_) / 256;              // 9216

  for (int l = 0; l < L_; ++l) {
    // ---- attention block ----
    ln_kernel<<<M_, 256, 0, stream>>>(x, ln1g + l * D_, ln1b + l * D_, hbf);

    convT_kernel<<<convBlocksDD, 256, 0, stream>>>(wq + (size_t)l * D_ * D_, wt, D_, D_);
    gemm_kernel<<<gemmBlocksD, 256, 0, stream>>>(hbf, wt, bq + l * D_, nullptr, nullptr, qb,
                                                 M_, D_, D_, 0);
    convT_kernel<<<convBlocksDD, 256, 0, stream>>>(wk + (size_t)l * D_ * D_, wt, D_, D_);
    gemm_kernel<<<gemmBlocksD, 256, 0, stream>>>(hbf, wt, bk + l * D_, nullptr, nullptr, kb,
                                                 M_, D_, D_, 0);
    convT_kernel<<<convBlocksDD, 256, 0, stream>>>(wv + (size_t)l * D_ * D_, wt, D_, D_);
    gemm_kernel<<<gemmBlocksD, 256, 0, stream>>>(hbf, wt, bv + l * D_, nullptr, nullptr, vb,
                                                 M_, D_, D_, 0);

    vT_kernel<<<(B_ * H_ * DH_ * S_) / 256, 256, 0, stream>>>(vb, vtb);
    rel_kernel<<<(B_ * H_ * S_ * R_) / 256, 256, 0, stream>>>(
        qb, rel_emb + (size_t)l * R_ * H_ * DH_, relb);

    attn_kernel<<<(B_ * H_ * (S_ / 16)) / 8, 256, 0, stream>>>(qb, kb, vtb, relb, relids,
                                                               amask, ctxb);

    convT_kernel<<<convBlocksDD, 256, 0, stream>>>(wo + (size_t)l * D_ * D_, wt, D_, D_);
    gemm_kernel<<<gemmBlocksD, 256, 0, stream>>>(ctxb, wt, bo + l * D_, x, x, nullptr,
                                                 M_, D_, D_, 0);

    // ---- FFN block ----
    ln_kernel<<<M_, 256, 0, stream>>>(x, ln2g + l * D_, ln2b + l * D_, hbf);

    convT_kernel<<<convBlocksDF, 256, 0, stream>>>(w1 + (size_t)l * D_ * F_, wt, D_, F_);
    gemm_kernel<<<gemmBlocksF, 256, 0, stream>>>(hbf, wt, b1 + l * F_, nullptr, nullptr, a1b,
                                                 M_, F_, D_, 1);
    convT_kernel<<<convBlocksDF, 256, 0, stream>>>(w2 + (size_t)l * F_ * D_, wt, F_, D_);
    gemm_kernel<<<gemmBlocksD, 256, 0, stream>>>(a1b, wt, b2 + l * D_, x, x, nullptr,
                                                 M_, D_, F_, 0);
  }
}
